// SpatioTemporalAttentionBlock_23922967838795
// MI455X (gfx1250) — matrix operand order repaired and checked by reading, not compiled, not run on hardware
//
#include <hip/hip_runtime.h>
#include <hip/hip_bf16.h>
#include <stddef.h>

// Problem constants
#define B_ 16
#define C_ 24
#define S_ 128
#define E_ 256
#define H_ 8
#define P_ 32

typedef __attribute__((ext_vector_type(16))) __bf16 v16bf;
typedef __attribute__((ext_vector_type(8)))  float  v8f;

union Frag { v16bf v; unsigned u[8]; };

__device__ inline unsigned short f2bf(float f) {
  unsigned u = __float_as_uint(f);
  unsigned r = u + 0x7FFFu + ((u >> 16) & 1u);   // round-to-nearest-even
  return (unsigned short)(r >> 16);
}
__device__ inline float bf2f(unsigned short u) {
  return __uint_as_float((unsigned)u << 16);
}

__device__ inline v8f wmma_bf16(const Frag& a, const Frag& b, v8f c) {
  return __builtin_amdgcn_wmma_f32_16x16x32_bf16(false, a.v, false, b.v,
                                                 (short)0, c, false, false);
}

// 16-bit A matrix (16x32) K offset for packed pair q (0..7); hi = 0 (lanes 0-15) or 8.
__device__ inline int a_koff(int q, int hi) {
  return ((q < 4) ? 2 * q : 16 + 2 * (q - 4)) + hi;
}

// Async copy of 16 bytes global -> LDS (ASYNCcnt-tracked, no VGPR data).
__device__ inline void async_b128(unsigned lds_byte_off, const void* gaddr) {
  unsigned long long ga = (unsigned long long)gaddr;
  asm volatile("global_load_async_to_lds_b128 %0, %1, off"
               :: "v"(lds_byte_off), "v"(ga) : "memory");
}
__device__ inline void wait_async0() {
  asm volatile("s_wait_asynccnt 0x0" ::: "memory");
}
// Wait until at most 3 async ops in flight (== one prefetch stage outstanding).
__device__ inline void wait_async3() {
  asm volatile("s_wait_asynccnt 0x3" ::: "memory");
}

// ---------------------------------------------------------------------------
// Weight packing kernels: f32 -> bf16, gathered into N-major [c][f][e] layout
// so GEMM B tiles are contiguous 64B rows (async-copyable, no transpose).
// ---------------------------------------------------------------------------
__global__ void pack_qs_kernel(const float* __restrict__ W, unsigned short* __restrict__ A) {
  // Wq_s[h][c][p][e] -> A[c][f][e], f = h*32+p
  int idx = blockIdx.x * 256 + threadIdx.x;
  if (idx >= C_ * E_ * E_) return;
  int e = idx % E_; int f = (idx / E_) % E_; int c = idx / (E_ * E_);
  int h = f >> 5, p = f & 31;
  A[idx] = f2bf(W[((size_t)(h * C_ + c) * P_ + p) * E_ + e]);
}

__global__ void pack_kvs_kernel(const float* __restrict__ W, unsigned short* __restrict__ A) {
  // Wk_s[h][e][p] -> A[f][e], f = h*32+p
  int idx = blockIdx.x * 256 + threadIdx.x;
  if (idx >= E_ * E_) return;
  int e = idx % E_; int f = idx / E_;
  int h = f >> 5, p = f & 31;
  A[idx] = f2bf(W[((size_t)h * E_ + e) * P_ + p]);
}

__global__ void pack_t_kernel(const float* __restrict__ W, unsigned short* __restrict__ A) {
  // W[h][c][e][p] -> A[c][f][e], f = h*32+p
  int idx = blockIdx.x * 256 + threadIdx.x;
  if (idx >= C_ * E_ * E_) return;
  int e = idx % E_; int f = (idx / E_) % E_; int c = idx / (E_ * E_);
  int h = f >> 5, p = f & 31;
  A[idx] = f2bf(W[((size_t)(h * C_ + c) * E_ + e) * P_ + p]);
}

__global__ void pack_cef_kernel(const float* __restrict__ W, unsigned short* __restrict__ A) {
  // W[c][e][f] -> A[c][f][e]
  int idx = blockIdx.x * 256 + threadIdx.x;
  if (idx >= C_ * E_ * E_) return;
  int e = idx % E_; int f = (idx / E_) % E_; int c = idx / (E_ * E_);
  A[idx] = f2bf(W[((size_t)c * E_ + e) * E_ + f]);
}

// f32 -> bf16 activation conversion
__global__ void cvt_bf16_kernel(const float* __restrict__ X, unsigned short* __restrict__ Y, int n) {
  int idx = blockIdx.x * 256 + threadIdx.x;
  if (idx < n) Y[idx] = f2bf(X[idx]);
}

// ---------------------------------------------------------------------------
// WMMA GEMM: Y[b,c,s,n] = sum_e X[b,c,s,e] * W[c][n][e] (+bias)(+relu)
// X and W are bf16. Double-buffered LDS staged with async b128 copies; the
// partial wait (asynccnt <= 3) overlaps stage kb+1's HBM traffic with stage
// kb's WMMAs. Block: 256 threads (8 waves). BM=128 (=S), BN=64, K-step 32.
// grid = (E/64=4, B, C). LDS rows 80B (16B-aligned, conflict-free stride).
// ---------------------------------------------------------------------------
__device__ inline void gemm_stage(const unsigned short* Xb, const unsigned short* WcN,
                                  int kb, unsigned ldsA, unsigned ldsB, int tid) {
  #pragma unroll
  for (int i = 0; i < 2; ++i) {
    int j = tid + i * 256;
    int row = j >> 2, ch = (j & 3) * 16;
    async_b128(ldsA + row * 80 + ch,
               (const char*)(Xb + (size_t)row * E_ + kb * 32) + ch);
  }
  int n = tid >> 2, ch = (tid & 3) * 16;
  async_b128(ldsB + n * 80 + ch,
             (const char*)(WcN + (size_t)n * E_ + kb * 32) + ch);
}

__global__ __launch_bounds__(256) void gemm_e256_kernel(
    const unsigned short* __restrict__ X, const unsigned short* __restrict__ W,
    const float* __restrict__ bias, float* __restrict__ Yf,
    unsigned short* __restrict__ Ybf, int wStrideC, int doRelu) {
  const int c = blockIdx.z, b = blockIdx.y;
  const int n0blk = blockIdx.x * 64;
  const int tid = threadIdx.x;
  const int lane = tid & 31, w = tid >> 5;
  const int mw = (w & 3) * 32;      // wave M offset (4 waves along M)
  const int nw = (w >> 2) * 32;     // wave N offset (2 waves along N)

  __shared__ unsigned short sA[2][128 * 40];  // [m][k], 80B rows, double buffered
  __shared__ unsigned short sB[2][64 * 40];   // [n][k], 80B rows, double buffered

  const unsigned short* Xb = X + (size_t)(b * C_ + c) * S_ * E_;
  const unsigned short* WcN = W + (size_t)c * wStrideC + (size_t)n0blk * E_;

  const unsigned ldsA0 = (unsigned)(unsigned long long)(const void*)sA[0];
  const unsigned ldsB0 = (unsigned)(unsigned long long)(const void*)sB[0];
  const unsigned ldsAoff[2] = {ldsA0, ldsA0 + 128 * 40 * 2};
  const unsigned ldsBoff[2] = {ldsB0, ldsB0 + 64 * 40 * 2};

  v8f acc[2][2] = {};
  const int msel = lane & 15;
  const int ahsel = (lane < 16) ? 0 : 8;
  const int bkh = (lane < 16) ? 0 : 16;

  gemm_stage(Xb, WcN, 0, ldsAoff[0], ldsBoff[0], tid);
  gemm_stage(Xb, WcN, 1, ldsAoff[1], ldsBoff[1], tid);

  for (int kb = 0; kb < 8; ++kb) {
    if (kb < 7) wait_async3(); else wait_async0();
    __syncthreads();  // stage kb valid in buf[kb&1] for all waves

    const unsigned short* cA = sA[kb & 1];
    const unsigned short* cB = sB[kb & 1];
    Frag a[2], bfr[2];
    #pragma unroll
    for (int i = 0; i < 2; ++i) {
      int m = mw + i * 16 + msel;
      #pragma unroll
      for (int q = 0; q < 8; ++q)
        a[i].u[q] = *(const unsigned*)&cA[m * 40 + a_koff(q, ahsel)];
    }
    #pragma unroll
    for (int j = 0; j < 2; ++j) {
      int n = nw + j * 16 + msel;
      #pragma unroll
      for (int q = 0; q < 8; ++q)
        bfr[j].u[q] = *(const unsigned*)&cB[n * 40 + a_koff(q, ahsel)];
    }
    #pragma unroll
    for (int i = 0; i < 2; ++i)
      #pragma unroll
      for (int j = 0; j < 2; ++j)
        acc[i][j] = wmma_bf16(a[i], bfr[j], acc[i][j]);

    __syncthreads();  // all waves done reading buf[kb&1]
    if (kb < 6) gemm_stage(Xb, WcN, kb + 2, ldsAoff[kb & 1], ldsBoff[kb & 1], tid);
  }

  const size_t obase = (size_t)(b * C_ + c) * S_ * E_;
  #pragma unroll
  for (int i = 0; i < 2; ++i) {
    int mbase = mw + i * 16 + ahsel;
    #pragma unroll
    for (int j = 0; j < 2; ++j) {
      int n = n0blk + nw + j * 16 + msel;
      float bv = bias ? bias[c * E_ + n] : 0.f;
      #pragma unroll
      for (int r = 0; r < 8; ++r) {
        float vv = acc[i][j][r] + bv;
        if (doRelu) vv = vv > 0.f ? vv : 0.f;
        size_t off = obase + (size_t)(mbase + r) * E_ + n;
        if (Yf) Yf[off] = vv;
        if (Ybf) Ybf[off] = f2bf(vv);
      }
    }
  }
}

// ---------------------------------------------------------------------------
// Temporal attention: per (b,h,c) 128x128 attention over time, WMMA both GEMMs.
// Q/K/V arrive as bf16; Q and K tiles staged with async b128 copies (rows are
// contiguous 64B), V copied synchronously (needs transpose). grid = (C, H, B),
// block 256 (8 waves, one 16-row slab each). Output bf16 for Wo_t GEMM.
// ---------------------------------------------------------------------------
__global__ __launch_bounds__(256) void temporal_attn_kernel(
    const unsigned short* __restrict__ Q, const unsigned short* __restrict__ K,
    const unsigned short* __restrict__ V, unsigned short* __restrict__ O) {
  const int c = blockIdx.x, h = blockIdx.y, b = blockIdx.z;
  const int tid = threadIdx.x, lane = tid & 31, w = tid >> 5;
  const size_t base = (size_t)(b * C_ + c) * S_ * E_ + h * 32;

  __shared__ unsigned short sQ[128 * 40];   // [s][p], 80B rows
  __shared__ unsigned short sK[128 * 40];   // [t][p] (== N-major B for QK^T)
  __shared__ unsigned short sVt[32 * 130];  // [p][t] (N-major B for P*V)
  __shared__ unsigned short sP[128 * 132];  // [s][t] probabilities (bf16)

  const unsigned short* Qb = Q + base;
  const unsigned short* Kb = K + base;
  const unsigned short* Vb = V + base;
  const unsigned ldsQ0 = (unsigned)(unsigned long long)(const void*)sQ;
  const unsigned ldsK0 = (unsigned)(unsigned long long)(const void*)sK;

  // Async-stage Q, K (512 x b128 each; 2 per thread per matrix)
  #pragma unroll
  for (int i = 0; i < 2; ++i) {
    int j = tid + i * 256;
    int row = j >> 2, ch = (j & 3) * 16;
    async_b128(ldsQ0 + row * 80 + ch, (const char*)(Qb + (size_t)row * E_) + ch);
    async_b128(ldsK0 + row * 80 + ch, (const char*)(Kb + (size_t)row * E_) + ch);
  }
  // V: plain copy with transpose into [p][t]
  {
    int row = tid >> 1, half = (tid & 1) * 16;
    const unsigned short* v = Vb + (size_t)row * E_ + half;
    #pragma unroll
    for (int i = 0; i < 16; ++i) sVt[(half + i) * 130 + row] = v[i];
  }
  wait_async0();
  __syncthreads();

  const int m0 = w * 16;
  const int msel = lane & 15;
  const int hsel8 = (lane < 16) ? 0 : 8;
  const int kh16 = (lane < 16) ? 0 : 16;

  // Scores = Q K^T  (one A-frag reused across 8 N tiles)
  Frag aq;
  #pragma unroll
  for (int q8 = 0; q8 < 8; ++q8)
    aq.u[q8] = *(const unsigned*)&sQ[(m0 + msel) * 40 + a_koff(q8, hsel8)];

  v8f sc[8];
  #pragma unroll
  for (int nt = 0; nt < 8; ++nt) {
    Frag bk;
    int n = nt * 16 + msel;
    #pragma unroll
    for (int q8 = 0; q8 < 8; ++q8)
      bk.u[q8] = *(const unsigned*)&sK[n * 40 + a_koff(q8, hsel8)];
    v8f z = {};
    sc[nt] = wmma_bf16(aq, bk, z);
  }

  // Row softmax in registers (reduce across 16-lane halves; rows split by half)
  const float scale = 0.17677669529663688f;  // 1/sqrt(32)
  #pragma unroll
  for (int i = 0; i < 8; ++i) {
    float mx = -1e30f;
    #pragma unroll
    for (int nt = 0; nt < 8; ++nt) { sc[nt][i] *= scale; mx = fmaxf(mx, sc[nt][i]); }
    #pragma unroll
    for (int msk = 1; msk < 16; msk <<= 1) mx = fmaxf(mx, __shfl_xor(mx, msk, 32));
    float sum = 0.f;
    #pragma unroll
    for (int nt = 0; nt < 8; ++nt) { float e = __expf(sc[nt][i] - mx); sc[nt][i] = e; sum += e; }
    #pragma unroll
    for (int msk = 1; msk < 16; msk <<= 1) sum += __shfl_xor(sum, msk, 32);
    float inv = 1.0f / sum;
    int row = m0 + hsel8 + i;
    #pragma unroll
    for (int nt = 0; nt < 8; ++nt) {
      sc[nt][i] *= inv;
      sP[row * 132 + nt * 16 + msel] = f2bf(sc[nt][i]);
    }
  }
  __syncthreads();

  // O = P V  (K = 128 in 4 steps, N = 32 in 2 tiles)
  v8f ob[2] = {};
  #pragma unroll
  for (int kk = 0; kk < 4; ++kk) {
    Frag ap;
    #pragma unroll
    for (int q8 = 0; q8 < 8; ++q8)
      ap.u[q8] = *(const unsigned*)&sP[(m0 + msel) * 132 + kk * 32 + a_koff(q8, hsel8)];
    #pragma unroll
    for (int j = 0; j < 2; ++j) {
      Frag bv;
      int n = j * 16 + msel;
      #pragma unroll
      for (int q8 = 0; q8 < 8; ++q8)
        bv.u[q8] = *(const unsigned*)&sVt[n * 130 + kk * 32 + a_koff(q8, hsel8)];
      ob[j] = wmma_bf16(ap, bv, ob[j]);
    }
  }

  unsigned short* Ob = O + base;
  #pragma unroll
  for (int j = 0; j < 2; ++j) {
    int n = j * 16 + msel;
    #pragma unroll
    for (int r = 0; r < 8; ++r) {
      int s = m0 + hsel8 + r;
      Ob[(size_t)s * E_ + n] = f2bf(ob[j][r]);
    }
  }
}

// ---------------------------------------------------------------------------
// Spatial attention: per (b,s,h) 24x24 attention over joints; scalar (1.7% of FLOPs)
// bf16 in / bf16 out (consumers: Wo_s projection GEMM).
// ---------------------------------------------------------------------------
__global__ __launch_bounds__(256) void spatial_attn_kernel(
    const unsigned short* __restrict__ Q, const unsigned short* __restrict__ K,
    const unsigned short* __restrict__ V, unsigned short* __restrict__ O) {
  int gid = blockIdx.x * 256 + threadIdx.x;
  if (gid >= B_ * S_ * H_ * C_) return;
  int c = gid % C_; int t = gid / C_;
  int h = t % H_; t /= H_;
  int s = t % S_; int b = t / S_;

  const size_t rowQ = ((size_t)(b * C_ + c) * S_ + s) * E_ + h * 32;
  float q[32];
  #pragma unroll
  for (int p = 0; p < 32; ++p) q[p] = bf2f(Q[rowQ + p]);

  const float scale = 0.17677669529663688f;
  float sc[24]; float mx = -1e30f;
  #pragma unroll
  for (int d = 0; d < C_; ++d) {
    size_t rk = ((size_t)(b * C_ + d) * S_ + s) * E_ + h * 32;
    float acc = 0.f;
    #pragma unroll
    for (int p = 0; p < 32; ++p) acc += q[p] * bf2f(K[rk + p]);
    acc *= scale;
    sc[d] = acc; mx = fmaxf(mx, acc);
  }
  float sum = 0.f;
  #pragma unroll
  for (int d = 0; d < C_; ++d) { sc[d] = __expf(sc[d] - mx); sum += sc[d]; }
  float inv = 1.0f / sum;
  float o[32] = {};
  #pragma unroll
  for (int d = 0; d < C_; ++d) {
    size_t rv = ((size_t)(b * C_ + d) * S_ + s) * E_ + h * 32;
    float pd = sc[d] * inv;
    #pragma unroll
    for (int p = 0; p < 32; ++p) o[p] += pd * bf2f(V[rv + p]);
  }
  #pragma unroll
  for (int p = 0; p < 32; ++p) O[rowQ + p] = f2bf(o[p]);
}

// ---------------------------------------------------------------------------
// LayerNorm helpers: one wave per E=256 row
// ---------------------------------------------------------------------------
__device__ inline void ln_row(float (&v)[8], int lane,
                              const float* __restrict__ g, const float* __restrict__ be) {
  float s = 0.f;
  #pragma unroll
  for (int i = 0; i < 8; ++i) s += v[i];
  #pragma unroll
  for (int m = 1; m < 32; m <<= 1) s += __shfl_xor(s, m, 32);
  float mean = s * (1.0f / 256.0f);
  float var = 0.f;
  #pragma unroll
  for (int i = 0; i < 8; ++i) { float d = v[i] - mean; var += d * d; }
  #pragma unroll
  for (int m = 1; m < 32; m <<= 1) var += __shfl_xor(var, m, 32);
  var *= (1.0f / 256.0f);
  float r = rsqrtf(var + 1e-6f);
  #pragma unroll
  for (int i = 0; i < 8; ++i) {
    int e = lane + i * 32;
    v[i] = (v[i] - mean) * r * g[e] + be[e];
  }
}

// attn_out = LN(x+spat) + LN(x+temp); emit both f32 (for final residual) and bf16 (for FF GEMM)
__global__ __launch_bounds__(256) void ln_add2_kernel(
    const float* __restrict__ x, const float* __restrict__ sp, const float* __restrict__ tp,
    const float* __restrict__ g, const float* __restrict__ be,
    float* __restrict__ outf, unsigned short* __restrict__ outbf) {
  int w = threadIdx.x >> 5, lane = threadIdx.x & 31;
  size_t row = (size_t)blockIdx.x * 8 + w;
  const float* xr = x + row * E_;
  const float* sr = sp + row * E_;
  const float* tr = tp + row * E_;
  float a[8], b2[8];
  #pragma unroll
  for (int i = 0; i < 8; ++i) {
    int e = lane + i * 32;
    float xv = xr[e];
    a[i] = xv + sr[e];
    b2[i] = xv + tr[e];
  }
  ln_row(a, lane, g, be);
  ln_row(b2, lane, g, be);
  #pragma unroll
  for (int i = 0; i < 8; ++i) {
    float vv = a[i] + b2[i];
    size_t off = row * E_ + lane + i * 32;
    outf[off] = vv;
    outbf[off] = f2bf(vv);
  }
}

__global__ __launch_bounds__(256) void ln_final_kernel(
    const float* __restrict__ attn, const float* __restrict__ ff,
    const float* __restrict__ g, const float* __restrict__ be, float* __restrict__ out) {
  int w = threadIdx.x >> 5, lane = threadIdx.x & 31;
  size_t row = (size_t)blockIdx.x * 8 + w;
  const float* ar = attn + row * E_;
  const float* fr = ff + row * E_;
  float a[8];
  #pragma unroll
  for (int i = 0; i < 8; ++i) { int e = lane + i * 32; a[i] = ar[e] + fr[e]; }
  ln_row(a, lane, g, be);
  #pragma unroll
  for (int i = 0; i < 8; ++i) out[row * E_ + lane + i * 32] = a[i];
}

// ---------------------------------------------------------------------------
// Host orchestration
// ---------------------------------------------------------------------------
extern "C" void kernel_launch(void* const* d_in, const int* in_sizes, int n_in,
                              void* d_out, int out_size, void* d_ws, size_t ws_size,
                              hipStream_t stream) {
  const float* x    = (const float*)d_in[0];
  const float* Wq_s = (const float*)d_in[1];
  const float* Wk_s = (const float*)d_in[2];
  const float* Wv_s = (const float*)d_in[3];
  const float* Wo_s = (const float*)d_in[4];
  const float* Wq_t = (const float*)d_in[5];
  const float* Wk_t = (const float*)d_in[6];
  const float* Wv_t = (const float*)d_in[7];
  const float* Wo_t = (const float*)d_in[8];
  const float* ln_g = (const float*)d_in[9];
  const float* ln_b = (const float*)d_in[10];
  const float* ff_w1 = (const float*)d_in[11];
  const float* ff_b1 = (const float*)d_in[12];
  const float* ff_w2 = (const float*)d_in[13];
  const float* ff_b2 = (const float*)d_in[14];
  float* out = (float*)d_out;

  const size_t CEF = (size_t)C_ * E_ * E_;          // per-joint weight elems
  const size_t EF  = (size_t)E_ * E_;               // shared weight elems
  const size_t ACT = (size_t)B_ * C_ * S_ * E_;     // activation elems

  char* wsp = (char*)d_ws;
  auto take = [&](size_t bytes) -> char* {
    char* r = wsp;
    wsp += (bytes + 255) & ~(size_t)255;
    return r;
  };

  unsigned short* wAqs = (unsigned short*)take(CEF * 2);
  unsigned short* wAks = (unsigned short*)take(EF * 2);
  unsigned short* wAvs = (unsigned short*)take(EF * 2);
  unsigned short* wAqt = (unsigned short*)take(CEF * 2);
  unsigned short* wAkt = (unsigned short*)take(CEF * 2);
  unsigned short* wAvt = (unsigned short*)take(CEF * 2);
  unsigned short* wWos = (unsigned short*)take(CEF * 2);
  unsigned short* wWot = (unsigned short*)take(CEF * 2);
  unsigned short* wF1  = (unsigned short*)take(CEF * 2);
  unsigned short* wF2  = (unsigned short*)take(CEF * 2);
  unsigned short* xbf    = (unsigned short*)take(ACT * 2);
  unsigned short* bufQbf = (unsigned short*)take(ACT * 2);
  unsigned short* bufKbf = (unsigned short*)take(ACT * 2);
  unsigned short* bufVbf = (unsigned short*)take(ACT * 2);
  unsigned short* bufObf = (unsigned short*)take(ACT * 2);
  unsigned short* bufAbf = (unsigned short*)take(ACT * 2);
  unsigned short* bufHbf = (unsigned short*)take(ACT * 2);
  float* bufSpat = (float*)take(ACT * 4);
  float* bufTemp = (float*)take(ACT * 4);
  float* bufAttn = (float*)take(ACT * 4);

  const int packBlocks = (int)((CEF + 255) / 256);   // 6144
  const int actBlocks  = (int)((ACT + 255) / 256);   // 49152
  const dim3 gemmGrid(4, B_, C_);
  const int strideC = (int)EF;

  // 1) Pack weights to bf16 [c][n][e]; convert x to bf16 once
  pack_qs_kernel<<<packBlocks, 256, 0, stream>>>(Wq_s, wAqs);
  pack_kvs_kernel<<<(int)((EF + 255) / 256), 256, 0, stream>>>(Wk_s, wAks);
  pack_kvs_kernel<<<(int)((EF + 255) / 256), 256, 0, stream>>>(Wv_s, wAvs);
  pack_t_kernel<<<packBlocks, 256, 0, stream>>>(Wq_t, wAqt);
  pack_t_kernel<<<packBlocks, 256, 0, stream>>>(Wk_t, wAkt);
  pack_t_kernel<<<packBlocks, 256, 0, stream>>>(Wv_t, wAvt);
  pack_cef_kernel<<<packBlocks, 256, 0, stream>>>(Wo_s, wWos);
  pack_cef_kernel<<<packBlocks, 256, 0, stream>>>(Wo_t, wWot);
  pack_cef_kernel<<<packBlocks, 256, 0, stream>>>(ff_w1, wF1);
  pack_cef_kernel<<<packBlocks, 256, 0, stream>>>(ff_w2, wF2);
  cvt_bf16_kernel<<<actBlocks, 256, 0, stream>>>(x, xbf, (int)ACT);

  // 2) Spatial branch: q/k/v projections (bf16 out) -> attention -> output projection
  gemm_e256_kernel<<<gemmGrid, 256, 0, stream>>>(xbf, wAqs, nullptr, nullptr, bufQbf, strideC, 0);
  gemm_e256_kernel<<<gemmGrid, 256, 0, stream>>>(xbf, wAks, nullptr, nullptr, bufKbf, 0, 0);
  gemm_e256_kernel<<<gemmGrid, 256, 0, stream>>>(xbf, wAvs, nullptr, nullptr, bufVbf, 0, 0);
  spatial_attn_kernel<<<(B_ * S_ * H_ * C_) / 256, 256, 0, stream>>>(bufQbf, bufKbf, bufVbf, bufObf);
  gemm_e256_kernel<<<gemmGrid, 256, 0, stream>>>(bufObf, wWos, nullptr, bufSpat, nullptr, strideC, 0);

  // 3) Temporal branch
  gemm_e256_kernel<<<gemmGrid, 256, 0, stream>>>(xbf, wAqt, nullptr, nullptr, bufQbf, strideC, 0);
  gemm_e256_kernel<<<gemmGrid, 256, 0, stream>>>(xbf, wAkt, nullptr, nullptr, bufKbf, strideC, 0);
  gemm_e256_kernel<<<gemmGrid, 256, 0, stream>>>(xbf, wAvt, nullptr, nullptr, bufVbf, strideC, 0);
  temporal_attn_kernel<<<dim3(C_, H_, B_), 256, 0, stream>>>(bufQbf, bufKbf, bufVbf, bufObf);
  gemm_e256_kernel<<<gemmGrid, 256, 0, stream>>>(bufObf, wWot, nullptr, bufTemp, nullptr, strideC, 0);

  // 4) attn_out = LN(x+spat) + LN(x+temp)  (f32 + bf16 copies)
  const int lnBlocks = (B_ * C_ * S_) / 8;  // 6144
  ln_add2_kernel<<<lnBlocks, 256, 0, stream>>>(x, bufSpat, bufTemp, ln_g, ln_b, bufAttn, bufAbf);

  // 5) FF: relu(attn @ w1 + b1) -> bf16 hidden; @ w2 + b2 -> f32
  gemm_e256_kernel<<<gemmGrid, 256, 0, stream>>>(bufAbf, wF1, ff_b1, nullptr, bufHbf, strideC, 1);
  gemm_e256_kernel<<<gemmGrid, 256, 0, stream>>>(bufHbf, wF2, ff_b2, bufSpat, nullptr, strideC, 0);

  // 6) out = LN(attn_out + ff)
  ln_final_kernel<<<lnBlocks, 256, 0, stream>>>(bufAttn, bufSpat, ln_g, ln_b, out);
}
